// GRU_D_layer_75411035783283
// MI455X (gfx1250) — matrix operand order repaired and checked by reading, not compiled, not run on hardware
//
#include <hip/hip_runtime.h>

typedef _Float16 v16h __attribute__((ext_vector_type(16)));
typedef _Float16 v8h  __attribute__((ext_vector_type(8)));
typedef float    v8f  __attribute__((ext_vector_type(8)));

#define B_  256
#define T_  512
#define D_  128
#define H_  256

#define WMMA_F16(a, b, c) \
  __builtin_amdgcn_wmma_f32_16x16x32_f16(false, (a), false, (b), (short)0, (c), false, false)

// Build a 16x32 f16 A-fragment for one lane from an f16 row in LDS.
// A layout (16-bit, 16x32): lane l<16 -> row l, K = {k0..k0+7, k0+16..k0+23};
// lane l>=16 -> row l-16, K = {k0+8..k0+15, k0+24..k0+31}.
// Caller passes rowbase = &row[k0 + 8*(lane>>4)].  Two ds_load_b128 + shuffle.
__device__ __forceinline__ v16h frag_from_f16row(const _Float16* rowbase) {
  v8h lo = *(const v8h*)(rowbase);
  v8h hi = *(const v8h*)(rowbase + 16);
  return __builtin_shufflevector(lo, hi, 0, 1, 2, 3, 4, 5, 6, 7,
                                         8, 9, 10, 11, 12, 13, 14, 15);
}

__device__ __forceinline__ float fast_sigmoid(float x) {
  return 1.0f / (1.0f + __expf(-x));
}
__device__ __forceinline__ float fast_tanh(float x) {
  return 2.0f / (1.0f + __expf(-2.0f * x)) - 1.0f;
}

// Convert an f32 KxN row-major matrix into per-lane f16 B fragments.
// dst layout: [kt][nt][lane][16 halves]; lane = column (lane&15) of n-tile,
// lanes 0-15 carry K = kt*32 + {0..7, 16..23}, lanes 16-31 carry K = kt*32 + {8..15, 24..31}.
__global__ void pack_b_kernel(const float* __restrict__ src, _Float16* __restrict__ dst,
                              int K, int N) {
  int tile = blockIdx.x;            // kt * (N/16) + nt
  int lane = threadIdx.x;           // 0..31
  int NT   = N >> 4;
  int kt   = tile / NT;
  int nt   = tile - kt * NT;
  int g    = lane >> 4;
  int col  = nt * 16 + (lane & 15);
  int kbase = kt * 32 + g * 8;
  _Float16* d = dst + ((size_t)tile * 32 + lane) * 16;
  (void)K;
#pragma unroll
  for (int h = 0; h < 16; ++h)
    d[h] = (_Float16)src[(size_t)(kbase + (h & 7) + ((h >> 3) << 4)) * N + col];
}

// Phase A: XP fragments = x[b,t,:] @ W_m + bias_m   (m: 0=z, 1=r, 2=h)
// XP layout = WMMA C/D fragment layout: [t][m][btile][nt][lane][8] f32.
__global__ __launch_bounds__(256) void proj_kernel(
    const float* __restrict__ x,       // [B][T][D]
    const _Float16* __restrict__ Wp,   // packed [3][4][16][32][16h]
    const float* __restrict__ bz, const float* __restrict__ br,
    const float* __restrict__ bh,
    float* __restrict__ XPf)           // fragment layout
{
  __shared__ __align__(16) _Float16 Xs[16][D_];
  int tid   = threadIdx.x;
  int lane  = tid & 31;
  int wave  = tid >> 5;                // 0..7
  int bt    = blockIdx.x;              // 16 btiles * T
  int btile = bt & 15;
  int t     = bt >> 4;
  int b0    = btile * 16;

  for (int i = tid; i < 16 * D_; i += 256) {
    int r = i >> 7, c = i & (D_ - 1);
    Xs[r][c] = (_Float16)x[((size_t)(b0 + r) * T_ + t) * D_ + c];
  }
  __syncthreads();

  int row = lane & 15, g = lane >> 4;
  int col = lane & 15;
  v16h A[4];
#pragma unroll
  for (int kt = 0; kt < 4; ++kt)
    A[kt] = frag_from_f16row(&Xs[row][kt * 32 + 8 * g]);

  const v16h* W = (const v16h*)Wp;
#pragma unroll
  for (int q = 0; q < 6; ++q) {
    int jj = wave + q * 8;             // 0..47
    int m  = jj >> 4;
    int nt = jj & 15;
    v8f acc = {};
#pragma unroll
    for (int kt = 0; kt < 4; ++kt)
      acc = WMMA_F16(A[kt], W[((size_t)(m * 4 + kt) * 16 + nt) * 32 + lane], acc);
    const float* bp = (m == 0) ? bz : (m == 1) ? br : bh;
    float bv = bp[nt * 16 + col];
#pragma unroll
    for (int v = 0; v < 8; ++v) acc[v] += bv;
    // fragment store: one contiguous 32B store per lane
    v8f* op = (v8f*)XPf + ((((size_t)t * 3 + m) * 16 + btile) * 16 + nt) * 32 + lane;
    *op = acc;
  }
}

// Phase B: sequential scan. One workgroup per 16 batch rows, 16 waves:
// wave w owns output n-tile w for all three recurrent GEMMs.  Gate values and
// h_dec are carried in registers (epilogue thread ownership is identical for
// all three gates); only the f16 GEMM-A operands (HD16, RH16) live in LDS.
__global__ __launch_bounds__(512) void scan_kernel(
    const float* __restrict__ h_decay, // [B][T]
    const _Float16* __restrict__ Up,   // packed [3][8][16][32][16h]
    const float* __restrict__ XPf,     // fragment layout [t][m][btile][nt][lane][8]
    float* __restrict__ out)           // [B][T][H]
{
  __shared__ __align__(16) _Float16 HD16[16][H_];   //  8 KB  dec * h   (GEMM A)
  __shared__ __align__(16) _Float16 RH16[16][H_];   //  8 KB  r*dec*h   (GEMM A)
  __shared__ __align__(16) float    Dec[16][T_];    // 32 KB  h_decay slice

  int tid  = threadIdx.x;
  int lane = tid & 31;
  int wave = tid >> 5;                 // 0..15
  int bidx = blockIdx.x;
  int b0   = bidx * 16;
  int row  = lane & 15, g = lane >> 4;
  int Mg   = 8 * g, col = lane & 15;
  int nt   = wave;
  int c    = nt * 16 + col;

  // stage decay slice + zero initial dec*h
  for (int i = tid; i < 16 * T_; i += 512) {
    int r = i >> 9, cc = i & (T_ - 1);
    Dec[r][cc] = h_decay[(size_t)(b0 + r) * T_ + cc];
  }
  for (int i = tid; i < 16 * H_ / 2; i += 512) ((float*)HD16)[i] = 0.0f;
  __syncthreads();

  const v16h* U  = (const v16h*)Up;
  const v8f*  XF = (const v8f*)XPf;

  // candidate-GEMM weights (Uh) stay register-resident: tail of the dep chain
  v16h Uhreg[8];
#pragma unroll
  for (int kt = 0; kt < 8; ++kt)
    Uhreg[kt] = U[((size_t)(2 * 8 + kt) * 16 + nt) * 32 + lane];

  float hdreg[8];                      // dec_t * h_{t-1} for owned elements
#pragma unroll
  for (int v = 0; v < 8; ++v) hdreg[v] = 0.0f;

  for (int t = 0; t < T_; ++t) {
    size_t fb = (((size_t)t * 3) * 16 + bidx) * 16 + nt;

    // prefetch next step's XP fragments (global_prefetch_b8)
    if (t + 1 < T_) {
      size_t nb = fb + (size_t)3 * 256;
      __builtin_prefetch((const float*)(XF + (nb + 0 * 256) * 32 + lane), 0, 1);
      __builtin_prefetch((const float*)(XF + (nb + 1 * 256) * 32 + lane), 0, 1);
      __builtin_prefetch((const float*)(XF + (nb + 2 * 256) * 32 + lane), 0, 1);
    }

    // input projections for this step (contiguous 32B fragment loads)
    v8f xpz = XF[(fb + 0 * 256) * 32 + lane];
    v8f xpr = XF[(fb + 1 * 256) * 32 + lane];
    v8f xph = XF[(fb + 2 * 256) * 32 + lane];

    // A fragments of dec*h
    v16h A[8];
#pragma unroll
    for (int kt = 0; kt < 8; ++kt)
      A[kt] = frag_from_f16row(&HD16[row][kt * 32 + 8 * g]);

    // z and r gate GEMMs (interleaved chains for ILP)
    v8f zacc = {}, racc = {};
#pragma unroll
    for (int kt = 0; kt < 8; ++kt) {
      zacc = WMMA_F16(A[kt], U[((size_t)(0 * 8 + kt) * 16 + nt) * 32 + lane], zacc);
      racc = WMMA_F16(A[kt], U[((size_t)(1 * 8 + kt) * 16 + nt) * 32 + lane], racc);
    }

    float zreg[8];
#pragma unroll
    for (int v = 0; v < 8; ++v) {
      float zv = fast_sigmoid(zacc[v] + xpz[v]);
      float rv = fast_sigmoid(racc[v] + xpr[v]);
      zreg[v] = zv;
      RH16[Mg + v][c] = (_Float16)(rv * hdreg[v]);
    }
    __syncthreads();

    // candidate GEMM on r * dec * h (register-resident Uh)
    v16h Ah[8];
#pragma unroll
    for (int kt = 0; kt < 8; ++kt)
      Ah[kt] = frag_from_f16row(&RH16[row][kt * 32 + 8 * g]);

    v8f hacc = {};
#pragma unroll
    for (int kt = 0; kt < 8; ++kt)
      hacc = WMMA_F16(Ah[kt], Uhreg[kt], hacc);

    // blend, emit h_t, and fuse next step's dec*h update
#pragma unroll
    for (int v = 0; v < 8; ++v) {
      int Mr = Mg + v;
      float hp = fast_tanh(hacc[v] + xph[v]);
      float zv = zreg[v];
      float hn = (1.0f - zv) * hdreg[v] + zv * hp;
      out[((size_t)(b0 + Mr) * T_ + t) * H_ + c] = hn;
      float dn  = (t + 1 < T_) ? Dec[Mr][t + 1] : 0.0f;
      float hd2 = dn * hn;
      hdreg[v] = hd2;
      HD16[Mr][c] = (_Float16)hd2;
    }
    __syncthreads();
  }
}

extern "C" void kernel_launch(void* const* d_in, const int* in_sizes, int n_in,
                              void* d_out, int out_size, void* d_ws, size_t ws_size,
                              hipStream_t stream) {
  (void)in_sizes; (void)n_in; (void)out_size; (void)ws_size;
  const float* x   = (const float*)d_in[0];
  const float* hdc = (const float*)d_in[1];
  const float* Wr  = (const float*)d_in[2];
  const float* Wz  = (const float*)d_in[3];
  const float* Wh  = (const float*)d_in[4];
  const float* Ur  = (const float*)d_in[5];
  const float* Uz  = (const float*)d_in[6];
  const float* Uh  = (const float*)d_in[7];
  const float* br  = (const float*)d_in[8];
  const float* bz  = (const float*)d_in[9];
  const float* bh  = (const float*)d_in[10];
  float* out = (float*)d_out;

  // Workspace layout:
  //   Wp:  3 * (128*256) f16  = 196608 B
  //   Up:  3 * (256*256) f16  = 393216 B
  //   XPf: 512*3*256*256 f32  = 402653184 B  (fragment layout)
  char* ws = (char*)d_ws;
  _Float16* Wp  = (_Float16*)ws;
  _Float16* Up  = (_Float16*)(ws + 196608);
  float*    XPf = (float*)(ws + 196608 + 393216);

  const size_t WSZ = (size_t)4 * 16 * 32 * 16;   // halves per packed W matrix
  const size_t USZ = (size_t)8 * 16 * 32 * 16;   // halves per packed U matrix

  // Pack weights (m order: 0=z, 1=r, 2=h)
  pack_b_kernel<<<4 * 16, 32, 0, stream>>>(Wz, Wp + 0 * WSZ, 128, 256);
  pack_b_kernel<<<4 * 16, 32, 0, stream>>>(Wr, Wp + 1 * WSZ, 128, 256);
  pack_b_kernel<<<4 * 16, 32, 0, stream>>>(Wh, Wp + 2 * WSZ, 128, 256);
  pack_b_kernel<<<8 * 16, 32, 0, stream>>>(Uz, Up + 0 * USZ, 256, 256);
  pack_b_kernel<<<8 * 16, 32, 0, stream>>>(Ur, Up + 1 * USZ, 256, 256);
  pack_b_kernel<<<8 * 16, 32, 0, stream>>>(Uh, Up + 2 * USZ, 256, 256);

  // Phase A: input projections for all timesteps (fully parallel)
  proj_kernel<<<16 * T_, 256, 0, stream>>>(x, Wp, bz, br, bh, XPf);

  // Phase B: sequential recurrence, 16 persistent workgroups
  scan_kernel<<<16, 512, 0, stream>>>(hdc, Up, XPf, out);
}
